// EdgeNetworkLayer_74852690034869
// MI455X (gfx1250) — matrix operand order repaired and checked by reading, not compiled, not run on hardware
//
#include <hip/hip_runtime.h>
#include <hip/hip_bf16.h>
#include <math.h>

#define Nn 8192
#define Hh 128
#define Ee 16384
#define EDd 16
#define HID 64

typedef __attribute__((ext_vector_type(16))) __bf16 v16bf;
typedef __attribute__((ext_vector_type(16))) unsigned short v16u;
typedef __attribute__((ext_vector_type(8))) float v8f;

__device__ __forceinline__ unsigned short f2bf(float f) {
    unsigned int u = __float_as_uint(f);
    unsigned int r = u + 0x7FFFu + ((u >> 16) & 1u);   // round-to-nearest-even
    return (unsigned short)(r >> 16);
}

// ---------------------------------------------------------------- zero
__global__ void k_zero(float* p, int n) {
    int i = blockIdx.x * 256 + threadIdx.x;
    if (i < n) p[i] = 0.f;
}

// ------------------------------------------------ edge MLP layer1 + A-frag pack
// hidden[e,k] = relu(ef[e,:]@W1[:,k] + b1[k]); pack 16-edge tiles into the
// CDNA5 bf16 A-matrix lane layout (ISA 7.12.2: lanes0-15 K=0-7,16-23; +8 hi half)
__global__ __launch_bounds__(64) void k_edge_mlp_pack(
        const float* __restrict__ ef, const float* __restrict__ W1,
        const float* __restrict__ b1, unsigned int* __restrict__ hfrag) {
    __shared__ float hid[16][65];
    int et = blockIdx.x, tid = threadIdx.x;          // tid = k index (0..63)
    for (int e = 0; e < 16; ++e) {
        const float* efe = ef + (size_t)(et * 16 + e) * EDd;
        float acc = b1[tid];
#pragma unroll
        for (int d = 0; d < EDd; ++d) acc += efe[d] * W1[d * HID + tid];
        hid[e][tid] = fmaxf(acc, 0.f);
    }
    __syncthreads();
    if (tid < 32) {
        int n = tid & 15, hi = tid >> 4;
        for (int s = 0; s < 2; ++s) {
            unsigned int* dst = hfrag + ((size_t)(et * 2 + s) * 32 + tid) * 8;
#pragma unroll
            for (int v = 0; v < 8; ++v) {
                int kl = (v < 4) ? 2 * v : 16 + 2 * (v - 4);
                int k0 = s * 32 + hi * 8 + kl;
                dst[v] = (unsigned int)f2bf(hid[n][k0]) |
                         ((unsigned int)f2bf(hid[n][k0 + 1]) << 16);
            }
        }
    }
}

// ---------------------------------------------------------------- gather h_w = h[src]
__global__ void k_gather(const float* __restrict__ h, const int* __restrict__ src,
                         float* __restrict__ hw) {
    int e = blockIdx.x, t = threadIdx.x;
    hw[(size_t)e * Hh + t] = h[(size_t)src[e] * Hh + t];
}

// ------------------------------------------- W2 -> bf16 B-fragment layout
// B-matrix 32x16: lanes0-15 hold K=0-7,16-23 (2/VGPR), lanes16-31 K=8-15,24-31.
__global__ __launch_bounds__(256) void k_pack_w2(const float* __restrict__ W2,
                                                 unsigned int* __restrict__ w2f) {
    int g = blockIdx.x * 256 + threadIdx.x;          // 1024*2*32*8 = 524288 dwords
    int v = g & 7, t = (g >> 3) & 31, s = (g >> 8) & 1, ct = g >> 9;
    int n = t & 15, hi = t >> 4;
    int k0 = s * 32 + hi * 8 + ((v < 4) ? 2 * v : 16 + 2 * (v - 4));
    int col = ct * 16 + n;
    float a = W2[(size_t)k0 * (Hh * Hh) + col];
    float b = W2[(size_t)(k0 + 1) * (Hh * Hh) + col];
    w2f[g] = (unsigned int)f2bf(a) | ((unsigned int)f2bf(b) << 16);
}

// ------------------------------------------- GRU weight [384,128] -> B-frag layout
// K=128 in 4 ksteps, 24 n-tiles; B[k][n] = W[nt*16+n][k]
__global__ __launch_bounds__(256) void k_pack_w3(const float* __restrict__ W,
                                                 unsigned int* __restrict__ wf) {
    int g = blockIdx.x * 256 + threadIdx.x;          // 24*4*32*8 = 24576 dwords
    if (g >= 24576) return;
    int v = g & 7, t = (g >> 3) & 31, s = (g >> 8) & 3, nt = g >> 10;
    int n = t & 15, hi = t >> 4;
    int k0 = s * 32 + hi * 8 + ((v < 4) ? 2 * v : 16 + 2 * (v - 4));
    const float* row = W + (size_t)(nt * 16 + n) * Hh;
    wf[g] = (unsigned int)f2bf(row[k0]) | ((unsigned int)f2bf(row[k0 + 1]) << 16);
}

// ------------------------------------------- fp32 rows [*,128] -> bf16 A-frag layout
// K=128 in 4 ksteps of 32; 16-row tiles
__global__ __launch_bounds__(256) void k_pack_rows(const float* __restrict__ X,
                                                   unsigned int* __restrict__ xf) {
    int g = blockIdx.x * 256 + threadIdx.x;          // 512*4*32*8 = 524288 dwords
    int v = g & 7, t = (g >> 3) & 31, s = (g >> 8) & 3, rt = g >> 10;
    int n = t & 15, hi = t >> 4;
    int kl = (v < 4) ? 2 * v : 16 + 2 * (v - 4);
    int k0 = s * 32 + hi * 8 + kl;
    const float* row = X + (size_t)(rt * 16 + n) * Hh;
    xf[g] = (unsigned int)f2bf(row[k0]) | ((unsigned int)f2bf(row[k0 + 1]) << 16);
}

// ------------------------------------------- fused A=hidden@W2+b2, matvec, scatter
// 32 edges (2 A-tiles) per WG: every B-fragment load feeds 4 WMMAs -> halves L2 traffic
__global__ __launch_bounds__(256) void k_edge_msg(
        const unsigned int* __restrict__ hfrag, const unsigned int* __restrict__ w2f,
        const float* __restrict__ hw, const float* __restrict__ b2,
        const int* __restrict__ tgt, float* __restrict__ m) {
    int et0 = blockIdx.x * 2;                  // two 16-edge tiles
    int lane = threadIdx.x & 31;
    int wave = threadIdx.x >> 5;               // 8 waves: wave w owns i = w*16..w*16+15
    int col = lane & 15, hi = lane >> 4;

    // A fragments for both edge tiles (K=0..31 and K=32..63)
    v16bf a0[2], a1[2];
#pragma unroll
    for (int u = 0; u < 2; ++u) {
        a0[u] = __builtin_bit_cast(v16bf,
            *(const v16u*)(hfrag + ((size_t)((et0 + u) * 2 + 0) * 32 + lane) * 8));
        a1[u] = __builtin_bit_cast(v16bf,
            *(const v16u*)(hfrag + ((size_t)((et0 + u) * 2 + 1) * 32 + lane) * 8));
    }

    // h_w values matching the C/D VGPR layout: reg r, lane -> (e=r+8*hi, j=t*16+col)
    float hwv[2][8][8];
#pragma unroll
    for (int u = 0; u < 2; ++u) {
        const float* hwb = hw + (size_t)(et0 + u) * 16 * Hh;
#pragma unroll
        for (int r = 0; r < 8; ++r)
#pragma unroll
            for (int t = 0; t < 8; ++t)
                hwv[u][r][t] = hwb[(r + 8 * hi) * Hh + t * 16 + col];
    }

    for (int ii = 0; ii < 16; ++ii) {
        int i = wave * 16 + ii;
        float part[2][8] = {{0, 0, 0, 0, 0, 0, 0, 0}, {0, 0, 0, 0, 0, 0, 0, 0}};
#pragma unroll
        for (int t = 0; t < 8; ++t) {          // j-tiles: j = t*16 + col
            size_t ct = (size_t)(i * 8 + t);
            v16bf b0 = __builtin_bit_cast(v16bf,
                *(const v16u*)(w2f + ((ct * 2 + 0) * 32 + lane) * 8));
            v16bf b1 = __builtin_bit_cast(v16bf,
                *(const v16u*)(w2f + ((ct * 2 + 1) * 32 + lane) * 8));
            float b2v = b2[i * Hh + t * 16 + col];   // fold +b2[c] exactly
#pragma unroll
            for (int u = 0; u < 2; ++u) {
                v8f acc = {0.f, 0.f, 0.f, 0.f, 0.f, 0.f, 0.f, 0.f};
                acc = __builtin_amdgcn_wmma_f32_16x16x32_bf16(false, a0[u], false, b0,
                                                              (short)0, acc, false, false);
                acc = __builtin_amdgcn_wmma_f32_16x16x32_bf16(false, a1[u], false, b1,
                                                              (short)0, acc, false, false);
#pragma unroll
                for (int r = 0; r < 8; ++r)
                    part[u][r] += (acc[r] + b2v) * hwv[u][r][t];
            }
        }
        // reduce over j (16 lanes within each half-wave), then scatter-add
#pragma unroll
        for (int u = 0; u < 2; ++u)
#pragma unroll
            for (int r = 0; r < 8; ++r) {
                float p = part[u][r];
                p += __shfl_xor(p, 1, 32);
                p += __shfl_xor(p, 2, 32);
                p += __shfl_xor(p, 4, 32);
                p += __shfl_xor(p, 8, 32);
                if (col == 0) {
                    int e = (et0 + u) * 16 + r + 8 * hi;
                    atomicAdd(&m[(size_t)tgt[e] * Hh + i], p);
                }
            }
    }
}

// ------------------------------------------- G = X @ W^T + bias via WMMA bf16
// xfrag: [512 row-tiles][4 ksteps][32][8]; wfrag: [24 n-tiles][4][32][8]
__global__ __launch_bounds__(256) void k_gemm_wmma(
        const unsigned int* __restrict__ xfrag, const unsigned int* __restrict__ wfrag,
        const float* __restrict__ bias, float* __restrict__ G) {
    int rt = blockIdx.x;                       // 512 row tiles
    int lane = threadIdx.x & 31;
    int wave = threadIdx.x >> 5;
    int col = lane & 15, hi = lane >> 4;
    v16bf a[4];
#pragma unroll
    for (int s = 0; s < 4; ++s)
        a[s] = __builtin_bit_cast(v16bf,
            *(const v16u*)(xfrag + ((size_t)(rt * 4 + s) * 32 + lane) * 8));
    for (int q = 0; q < 3; ++q) {
        int nt = wave * 3 + q;                 // 24 n-tiles of 16 (384 cols)
        v8f acc = {0.f, 0.f, 0.f, 0.f, 0.f, 0.f, 0.f, 0.f};
#pragma unroll
        for (int s = 0; s < 4; ++s) {
            v16bf b = __builtin_bit_cast(v16bf,
                *(const v16u*)(wfrag + ((size_t)(nt * 4 + s) * 32 + lane) * 8));
            acc = __builtin_amdgcn_wmma_f32_16x16x32_bf16(false, a[s], false, b,
                                                          (short)0, acc, false, false);
        }
        float bv = bias[nt * 16 + col];
#pragma unroll
        for (int r = 0; r < 8; ++r)
            G[(size_t)(rt * 16 + r + 8 * hi) * 384 + nt * 16 + col] = acc[r] + bv;
    }
}

// ---------------------------------------------------------------- GRU gates
__global__ void k_gru_gate(const float* __restrict__ GI, const float* __restrict__ GH,
                           const float* __restrict__ h, float* __restrict__ out) {
    int idx = blockIdx.x * 256 + threadIdx.x;        // N*H
    int row = idx >> 7, j = idx & 127;
    size_t b = (size_t)row * 384;
    float rr = GI[b + j] + GH[b + j];
    float zz = GI[b + 128 + j] + GH[b + 128 + j];
    float gin = GI[b + 256 + j], ghn = GH[b + 256 + j];
    rr = 1.f / (1.f + __expf(-rr));
    zz = 1.f / (1.f + __expf(-zz));
    float nn = tanhf(gin + rr * ghn);
    float hv = h[(size_t)row * Hh + j];
    out[idx] = (1.f - zz) * nn + zz * hv;
}

extern "C" void kernel_launch(void* const* d_in, const int* in_sizes, int n_in,
                              void* d_out, int out_size, void* d_ws, size_t ws_size,
                              hipStream_t stream) {
    (void)in_sizes; (void)n_in; (void)out_size; (void)ws_size;
    const float* h   = (const float*)d_in[0];
    const int*   ei  = (const int*)d_in[1];     // [2,E]: src=ei, tgt=ei+E
    const float* ef  = (const float*)d_in[2];
    const float* W1  = (const float*)d_in[3];
    const float* b1  = (const float*)d_in[4];
    const float* W2  = (const float*)d_in[5];
    const float* b2  = (const float*)d_in[6];
    const float* Wih = (const float*)d_in[7];
    const float* Whh = (const float*)d_in[8];
    const float* bih = (const float*)d_in[9];
    const float* bhh = (const float*)d_in[10];
    float* out = (float*)d_out;

    char* ws = (char*)d_ws;
    unsigned int* hfrag = (unsigned int*)(ws);                        // 2 MB
    unsigned int* w2f   = (unsigned int*)(ws + (2ull  << 20));        // 2 MB
    float*        hw    = (float*)      (ws + (4ull  << 20));         // 8 MB
    float*        m     = (float*)      (ws + (12ull << 20));         // 4 MB
    unsigned int* wihf  = (unsigned int*)(ws + (16ull << 20));        // 96 KB
    unsigned int* whhf  = (unsigned int*)(ws + (16ull << 20) + (128ull << 10)); // 96 KB
    unsigned int* hfN   = (unsigned int*)(ws + (17ull << 20));        // 2 MB
    unsigned int* mfrag = (unsigned int*)(ws + (19ull << 20));        // 2 MB
    float*        GH    = (float*)      (ws + (21ull << 20));         // 12 MB
    float*        GI    = (float*)      (ws);   // reuses staging region after k_edge_msg

    k_zero<<<(Nn * Hh + 255) / 256, 256, 0, stream>>>(m, Nn * Hh);
    k_edge_mlp_pack<<<Ee / 16, 64, 0, stream>>>(ef, W1, b1, hfrag);
    k_gather<<<Ee, Hh, 0, stream>>>(h, ei, hw);
    k_pack_w2<<<(1024 * 2 * 32 * 8) / 256, 256, 0, stream>>>(W2, w2f);
    k_pack_w3<<<96, 256, 0, stream>>>(Wih, wihf);
    k_pack_w3<<<96, 256, 0, stream>>>(Whh, whhf);
    k_pack_rows<<<2048, 256, 0, stream>>>(h, hfN);

    k_edge_msg<<<Ee / 32, 256, 0, stream>>>(hfrag, w2f, hw, b2, ei + Ee, m);

    k_pack_rows<<<2048, 256, 0, stream>>>(m, mfrag);
    k_gemm_wmma<<<Nn / 16, 256, 0, stream>>>(mfrag, wihf, bih, GI);
    k_gemm_wmma<<<Nn / 16, 256, 0, stream>>>(hfN, whhf, bhh, GH);
    k_gru_gate<<<(Nn * Hh) / 256, 256, 0, stream>>>(GI, GH, h, out);
}
